// LSTM_1511828488820
// MI455X (gfx1250) — matrix operand order repaired and checked by reading, not compiled, not run on hardware
//
#include <hip/hip_runtime.h>
#include <hip/hip_bf16.h>
#include <math.h>

// Problem sizes (fixed by the reference)
#define S_LEN 16384
#define E_DIM 512
#define H_DIM 512
#define G4H   2048   // 4*H

#define USE_ASYNC_H_STAGE 1

// ---------------- WMMA types ----------------
typedef __attribute__((ext_vector_type(16))) __bf16 v16bf;
typedef __attribute__((ext_vector_type(2)))  __bf16 v2bf;
typedef __attribute__((ext_vector_type(8)))  float  v8f;

union Frag { v16bf v; unsigned short s[16]; unsigned p[8]; uint4 q[2]; };

__device__ __forceinline__ unsigned pk2bf(float a, float b) {
#if __has_builtin(__builtin_amdgcn_cvt_pk_bf16_f32)
    return __builtin_bit_cast(unsigned, __builtin_amdgcn_cvt_pk_bf16_f32(a, b));
#else
    // fptrunc: backend lowers to the hw bf16 convert on gfx1250 (RNE)
    v2bf t; t.x = (__bf16)a; t.y = (__bf16)b;
    return __builtin_bit_cast(unsigned, t);
#endif
}
__device__ __forceinline__ float sigmoidf(float x) {
    return 1.0f / (1.0f + __expf(-x));
}

// =====================================================================
// Kernel 0: one-shot W_ih fp32 -> bf16 (2 MB copy in workspace).
// Removes per-tile conversion + halves B-side L2 traffic in the GEMM.
// =====================================================================
__global__ void __launch_bounds__(256)
cvt_wih_bf16(const float* __restrict__ W_ih, unsigned* __restrict__ wbf) {
    const size_t i = ((size_t)blockIdx.x * 256 + threadIdx.x) * 8;  // 8 floats
    const float4 f0 = *reinterpret_cast<const float4*>(&W_ih[i]);
    const float4 f1 = *reinterpret_cast<const float4*>(&W_ih[i + 4]);
    uint4 p;
    p.x = pk2bf(f0.x, f0.y); p.y = pk2bf(f0.z, f0.w);
    p.z = pk2bf(f1.x, f1.y); p.w = pk2bf(f1.z, f1.w);
    *reinterpret_cast<uint4*>(&wbf[i >> 1]) = p;
}

// =====================================================================
// Kernel 1: fused embedding gather + gx = emb[x] @ W_ih^T + (b_ih+b_hh)
// bf16 WMMA 16x16x32, fp32 accumulate.  WG tile: 64 (S) x 128 (4H).
// Grid: (16, 256), 256 threads (8 waves, 4x2 of 16x64 sub-tiles).
// Double-buffered LDS; next-tile global loads overlap current WMMAs.
// LDS tiles are fragment-ready (32 contiguous bf16 K per row/col).
// =====================================================================
#define TM 64
#define TN 128
#define TK 32
#define LPAD 8          // 80-byte row stride: 16B aligned, conflict-free

__global__ void __launch_bounds__(256)
embed_gemm_wmma(const int* __restrict__ x,
                const float* __restrict__ emb,
                const unsigned short* __restrict__ wbf,   // W_ih in bf16
                const float* __restrict__ b_ih,
                const float* __restrict__ b_hh,
                float* __restrict__ gx) {
    __shared__ int sidx[TM];
    __shared__ unsigned short As[2][TM][TK + LPAD];
    __shared__ unsigned short Bs[2][TN][TK + LPAD];

    const int tid  = threadIdx.x;
    const int m0   = blockIdx.y * TM;
    const int n0   = blockIdx.x * TN;
    const int lane = tid & 31;
    const int wid  = tid >> 5;
    const int wave_m = wid & 3;        // 4 tiles of 16 rows
    const int wave_n = wid >> 2;       // 2 strips of 64 cols
    const int half = lane >> 4;        // lane group (K-interleave select)
    const int l16  = lane & 15;

    if (tid < TM) sidx[tid] = x[m0 + tid];
    __syncthreads();

    v8f acc[4];
#pragma unroll
    for (int s = 0; s < 4; ++s) acc[s] = (v8f){0.f,0.f,0.f,0.f,0.f,0.f,0.f,0.f};

    // A-load role: row = tid/4, 8 consecutive K (fp32 gather + convert)
    const int arow = tid >> 2;
    const int acol = (tid & 3) * 8;
    const size_t aro = (size_t)sidx[arow] * E_DIM + acol;
    // B-load role: col n = tid/2, 16 consecutive K (pure bf16 copy)
    const int bn  = tid >> 1;
    const int bk0 = (tid & 1) * 16;
    const size_t bro = (size_t)(n0 + bn) * E_DIM + bk0;

    // ---- prologue: stage tile k0=0 into registers ----
    float4 a0 = *reinterpret_cast<const float4*>(&emb[aro]);
    float4 a1 = *reinterpret_cast<const float4*>(&emb[aro + 4]);
    uint4  bb0 = *reinterpret_cast<const uint4*>(&wbf[bro]);
    uint4  bb1 = *reinterpret_cast<const uint4*>(&wbf[bro + 8]);

    int buf = 0;
    for (int k0 = 0; k0 < E_DIM; k0 += TK) {
        // ---- staged regs -> LDS[buf] (A converts, B is a straight copy) ----
        {
            uint4 pa;
            pa.x = pk2bf(a0.x, a0.y); pa.y = pk2bf(a0.z, a0.w);
            pa.z = pk2bf(a1.x, a1.y); pa.w = pk2bf(a1.z, a1.w);
            *reinterpret_cast<uint4*>(&As[buf][arow][acol]) = pa;
            *reinterpret_cast<uint4*>(&Bs[buf][bn][bk0 + 0]) = bb0;
            *reinterpret_cast<uint4*>(&Bs[buf][bn][bk0 + 8]) = bb1;
        }
        __syncthreads();

        // ---- issue next tile's global loads (overlap with WMMAs below) ----
        if (k0 + TK < E_DIM) {
            const size_t an  = aro + k0 + TK;
            const size_t bnx = bro + k0 + TK;
            a0  = *reinterpret_cast<const float4*>(&emb[an]);
            a1  = *reinterpret_cast<const float4*>(&emb[an + 4]);
            bb0 = *reinterpret_cast<const uint4*>(&wbf[bnx]);
            bb1 = *reinterpret_cast<const uint4*>(&wbf[bnx + 8]);
        }

        // ---- A fragment (16-bit 16x32 ISA layout): lanes<16 K{0..7,16..23},
        //      lanes>=16 K{8..15,24..31} -> two contiguous 16B chunks ----
        Frag af;
        const int am = wave_m * 16 + l16;
        af.q[0] = *reinterpret_cast<const uint4*>(&As[buf][am][8 * half]);
        af.q[1] = *reinterpret_cast<const uint4*>(&As[buf][am][16 + 8 * half]);

        // ---- 4 B fragments + 4 WMMAs (loads pipeline past the matrix ops)
        Frag bfr[4];
#pragma unroll
        for (int sub = 0; sub < 4; ++sub) {
            const int bc = wave_n * 64 + sub * 16 + l16;
            bfr[sub].q[0] = *reinterpret_cast<const uint4*>(&Bs[buf][bc][8 * half]);
            bfr[sub].q[1] = *reinterpret_cast<const uint4*>(&Bs[buf][bc][16 + 8 * half]);
        }
#pragma unroll
        for (int sub = 0; sub < 4; ++sub)
            acc[sub] = __builtin_amdgcn_wmma_f32_16x16x32_bf16(
                false, af.v, false, bfr[sub].v, (short)0, acc[sub], false, false);

        buf ^= 1;
    }

    // ---- epilogue: add (b_ih + b_hh), store fp32 (C layout: M = r + 8*half) ----
#pragma unroll
    for (int sub = 0; sub < 4; ++sub) {
        const int col = n0 + wave_n * 64 + sub * 16 + l16;
        const float bias = b_ih[col] + b_hh[col];
#pragma unroll
        for (int r = 0; r < 8; ++r) {
            const int row = m0 + wave_m * 16 + 8 * half + r;
            gx[(size_t)row * G4H + col] = acc[sub][r] + bias;
        }
    }
}

// =====================================================================
// Kernel 2: persistent LSTM recurrence.  64 WGs x 256 threads.
// W_hh lives in VGPRs (64 f32/thread = entire 4 MB across the grid).
// Per step: h (2 KB) async-copied global->LDS, FMA matvec, wave-0
// reduction + gate math, device-scope phase barrier.
// =====================================================================
#define NWG 64

__global__ void __launch_bounds__(256)
lstm_scan(const float* __restrict__ gx,
          const float* __restrict__ W_hh,
          float* __restrict__ hglob,
          unsigned* __restrict__ bar) {   // bar[0]=count, bar[1]=phase
    __shared__ float hls[H_DIM];
    __shared__ float part[256];

    const int tid = threadIdx.x;
    const int wg  = blockIdx.x;
    const int r   = tid >> 3;                 // local gate-row 0..31
    const int kc  = (tid & 7) * 64;           // k-chunk of 64
    const int gate = r >> 3;                  // 0..3 (i,f,g,o)
    const int hloc = r & 7;                   // 0..7
    const int grow = gate * H_DIM + wg * 8 + hloc;   // row of W_hh

    // Resident weights
    float w[64];
#pragma unroll
    for (int i = 0; i < 64; ++i)
        w[i] = W_hh[(size_t)grow * H_DIM + kc + i];

#if USE_ASYNC_H_STAGE
    const unsigned hls_off = (unsigned)(size_t)(&hls[tid * 4]);  // LDS byte offset
    const float*   hsrc    = hglob + tid * 4;
#endif

    float c = 0.0f;   // cell state for lanes 0..7 of wave 0 (hloc = tid)

    for (int t = 0; t < S_LEN; ++t) {
        // ---- stage h (2 KB) into LDS ----
#if USE_ASYNC_H_STAGE
        if (tid < 128) {   // 128 lanes x b128 = 2 KB, no VGPR round-trip
            asm volatile("global_load_async_to_lds_b128 %0, %1, off"
                         :: "v"(hls_off), "v"(hsrc) : "memory");
        }
        asm volatile("s_wait_asynccnt 0" ::: "memory");
#else
        hls[tid]       = hglob[tid];
        hls[tid + 256] = hglob[tid + 256];
#endif
        __syncthreads();

        float p = 0.0f;
#pragma unroll
        for (int i = 0; i < 64; ++i)
            p = __builtin_fmaf(w[i], hls[kc + i], p);
        part[tid] = p;

        // stream next gx row toward the caches (global_prefetch_b8)
        if (tid < 64 && t + 1 < S_LEN)
            __builtin_prefetch(&gx[(size_t)(t + 1) * G4H + tid * 32], 0, 1);
        __syncthreads();

        if (tid < 32) {   // wave 0: reduce 8 partials per gate-row
            float s = 0.0f;
#pragma unroll
            for (int j = 0; j < 8; ++j) s += part[tid * 8 + j];
            const int gg = tid >> 3, hh = tid & 7;
            float g = s + gx[(size_t)t * G4H + gg * H_DIM + wg * 8 + hh];
            // gather i,f,g,o for my h index via wave32 shuffles
            const float gi = __shfl(g, hh,      32);
            const float gf = __shfl(g, hh + 8,  32);
            const float gc = __shfl(g, hh + 16, 32);
            const float go = __shfl(g, hh + 24, 32);
            if (tid < 8) {
                const float i_ = sigmoidf(gi);
                const float f_ = sigmoidf(gf);
                const float c_ = tanhf(gc);
                const float o_ = sigmoidf(go);
                c = __builtin_fmaf(f_, c, i_ * c_);
                hglob[wg * 8 + tid] = o_ * tanhf(c);
            }
        }

        // ---- grid-wide phase barrier (device scope) ----
        __threadfence();          // release my h slice to L2
        __syncthreads();
        if (tid == 0) {
            const unsigned a = __hip_atomic_fetch_add(&bar[0], 1u,
                                   __ATOMIC_ACQ_REL, __HIP_MEMORY_SCOPE_AGENT);
            if (a == NWG - 1) {
                bar[0] = 0;
                __hip_atomic_store(&bar[1], (unsigned)(t + 1),
                                   __ATOMIC_RELEASE, __HIP_MEMORY_SCOPE_AGENT);
            }
            while (__hip_atomic_load(&bar[1], __ATOMIC_ACQUIRE,
                                     __HIP_MEMORY_SCOPE_AGENT) < (unsigned)(t + 1))
                __builtin_amdgcn_s_sleep(1);
        }
        __syncthreads();
    }
}

// =====================================================================
// Kernel 3: tiny MLP head: sigmoid(sigmoid(h@W1^T + b1) @ W2^T + b2)
// =====================================================================
__global__ void __launch_bounds__(128)
mlp_head(const float* __restrict__ h,
         const float* __restrict__ W1, const float* __restrict__ b1,
         const float* __restrict__ W2, const float* __restrict__ b2,
         float* __restrict__ out) {
    __shared__ float hid[128];
    const int j = threadIdx.x;
    float a = b1[j];
    for (int k = 0; k < H_DIM; ++k)
        a = __builtin_fmaf(h[k], W1[(size_t)j * H_DIM + k], a);
    hid[j] = sigmoidf(a);
    __syncthreads();
    if (j < 9) {
        float o = b2[j];
#pragma unroll
        for (int k = 0; k < 128; ++k)
            o = __builtin_fmaf(hid[k], W2[j * 128 + k], o);
        out[j] = sigmoidf(o);
    }
}

// Reset h0/c0 and the grid barrier so graph replays are deterministic.
__global__ void init_state(float* __restrict__ hglob, unsigned* __restrict__ bar) {
    const int t = threadIdx.x;
    if (t < H_DIM) hglob[t] = 0.0f;
    if (t == 0) { bar[0] = 0u; bar[1] = 0u; }
}

// =====================================================================
extern "C" void kernel_launch(void* const* d_in, const int* in_sizes, int n_in,
                              void* d_out, int out_size, void* d_ws, size_t ws_size,
                              hipStream_t stream) {
    (void)in_sizes; (void)n_in; (void)out_size; (void)ws_size;
    const int*   x     = (const int*)  d_in[0];
    const float* emb   = (const float*)d_in[1];
    const float* W_ih  = (const float*)d_in[2];
    const float* W_hh  = (const float*)d_in[3];
    const float* b_ih  = (const float*)d_in[4];
    const float* b_hh  = (const float*)d_in[5];
    const float* W1    = (const float*)d_in[6];
    const float* b1    = (const float*)d_in[7];
    const float* W2    = (const float*)d_in[8];
    const float* b2    = (const float*)d_in[9];
    float* out = (float*)d_out;

    // workspace: gx[S,4H] fp32 (128 MiB) | h[512] | barrier[2] | pad | W_ih bf16 (2 MiB)
    const size_t GXB = (size_t)S_LEN * G4H * sizeof(float);
    float*    gx    = (float*)d_ws;
    float*    hglob = (float*)((char*)d_ws + GXB);
    unsigned* bar   = (unsigned*)(hglob + H_DIM);
    unsigned* wbf   = (unsigned*)((char*)d_ws + GXB + 4096);

    cvt_wih_bf16<<<(G4H * E_DIM / 8) / 256, 256, 0, stream>>>(W_ih, wbf);
    init_state<<<1, 512, 0, stream>>>(hglob, bar);
    embed_gemm_wmma<<<dim3(G4H / TN, S_LEN / TM), 256, 0, stream>>>(
        x, emb, (const unsigned short*)wbf, b_ih, b_hh, gx);
    lstm_scan<<<NWG, 256, 0, stream>>>(gx, W_hh, hglob, bar);
    mlp_head<<<1, 128, 0, stream>>>(hglob, W1, b1, W2, b2, out);
}
